// SchNetSpinLayer_38362647888618
// MI455X (gfx1250) — matrix operand order repaired and checked by reading, not compiled, not run on hardware
//
#include <hip/hip_runtime.h>
#include <stdint.h>

typedef __attribute__((ext_vector_type(16))) __bf16 v16bf;
typedef __attribute__((ext_vector_type(8)))  __bf16 v8bf;
typedef __attribute__((ext_vector_type(4)))  __bf16 v4bf;
typedef __attribute__((ext_vector_type(8)))  float  v8f;
typedef __attribute__((ext_vector_type(4)))  float  v4f;

#define B_    256
#define NELEC 64
#define NNUC  16
#define DIST  32
#define EMB   64
#define NUP   32

// ---------- helpers ----------

__device__ __forceinline__ float sspf(float z) {
  // SchNet shifted softplus: ln(1+e^z) - ln2 = ln2 * (log2(1 + exp2(z*log2e)) - 1)
  // v_exp_f32 / v_log_f32 are natively base-2 on CDNA5 -> 2 TRANS + 2 FMA, branch-free.
  float t  = __builtin_amdgcn_exp2f(z * 1.44269504088896340736f);
  float sp = 0.6931471805599453f * (__builtin_amdgcn_logf(1.0f + t) - 1.0f);
  return (z > 60.0f) ? (z - 0.6931471805599453f) : sp;  // overflow guard (cndmask)
}

__device__ __forceinline__ void load_f32(float* dst, const float* __restrict__ src,
                                         int n, int tid, int nt) {
  for (int i = tid; i < n; i += nt) dst[i] = src[i];
}

// Store a [K x 64] f32 weight matrix into WMMA-B fragment-major bf16 layout:
// unit u = (kc*4 + nt)*32 + lane holds the 16 contiguous halves of that lane's
// B fragment: element e = W[kc*32 + 16*(lane>>4) + e][nt*16 + (lane&15)].
__device__ __forceinline__ void load_w_swz(v16bf* dst, const float* __restrict__ src,
                                           int K, int tid, int nthreads) {
  int nunits = (K >> 5) * 128;  // (K/32) k-chunks * 4 n-tiles * 32 lanes
  for (int u = tid; u < nunits; u += nthreads) {
    int lane = u & 31, nt = (u >> 5) & 3, kc = u >> 7;
    const float* s = src + (long)(kc * 32 + ((lane >> 4) << 4)) * EMB + nt * 16 + (lane & 15);
    v16bf v;
#pragma unroll
    for (int e = 0; e < 16; ++e) v[e] = (__bf16)s[e * EMB];
    dst[u] = v;  // one 32-byte aligned LDS store
  }
}

// Vectorized f32 -> bf16 LDS staging (n multiple of 128; src 16B aligned)
__device__ __forceinline__ void stage_bf16(__bf16* dst, const float* __restrict__ src,
                                           int n, int lane) {
  for (int i = lane * 4; i < n; i += 128) {
    v4f s = *(const v4f*)(src + i);
    v4bf d;
#pragma unroll
    for (int e = 0; e < 4; ++e) d[e] = (__bf16)s[e];
    *(v4bf*)(dst + i) = d;
  }
}

// f32 -> f32 LDS copy via CDNA5 async DMA (tracked by ASYNCcnt, no VGPR transit).
// LDS byte offset = low 32 bits of the generic shared-memory address (ISA 10.2).
__device__ __forceinline__ void async_copy_f32(float* lds_dst, const float* __restrict__ src,
                                               int n, int lane) {
  for (int i = lane * 4; i < n; i += 128) {
    uint32_t dsaddr = (uint32_t)(uintptr_t)(lds_dst + i);
    const float* gaddr = src + i;
    asm volatile("global_load_async_to_lds_b128 %0, %1, off"
                 :: "v"(dsaddr), "v"(gaddr) : "memory");
  }
}
__device__ __forceinline__ void async_wait0() {
  asm volatile("s_wait_asynccnt 0" ::: "memory");
}

// A fragment: 16xK row-major bf16 tile; each lane's 16 elements are one
// contiguous 16-half run (k = kb + 16*hi + 0..15) -> two aligned 16-byte LDS loads.
__device__ __forceinline__ v16bf load_a_frag(const __bf16* A, int lda, int kb, int lane) {
  int m = lane & 15, hi = lane >> 4;
  const __bf16* p = A + m * lda + kb + 16 * hi;
  v8bf lo = *(const v8bf*)(p);       // k = kb + 16*hi + 0..7
  v8bf hs = *(const v8bf*)(p + 8);   // k = kb + 16*hi + 8..15
  v16bf a;
#pragma unroll
  for (int e = 0; e < 8; ++e) { a[e] = lo[e]; a[e + 8] = hs[e]; }
  return a;
}

// One wave: D[16,64] = A[16,K] * W[K,64] + bias   (acc[nt] covers N = nt*16..+15)
__device__ __forceinline__ void gemm16xK(const __bf16* A, int lda, const v16bf* Wf,
                                         const float* bias, int K, int lane, v8f acc[4]) {
#pragma unroll
  for (int nt = 0; nt < 4; ++nt) {
    float bv = bias[nt * 16 + (lane & 15)];
    v8f c;
#pragma unroll
    for (int r = 0; r < 8; ++r) c[r] = bv;
    acc[nt] = c;
  }
  for (int kb = 0, kc = 0; kb < K; kb += 32, ++kc) {
    v16bf a = load_a_frag(A, lda, kb, lane);
#pragma unroll
    for (int nt = 0; nt < 4; ++nt) {
      v16bf b = Wf[(kc * 4 + nt) * 32 + lane];  // fragment-major: 32B contiguous
      acc[nt] = __builtin_amdgcn_wmma_f32_16x16x32_bf16(false, a, false, b,
                                                        (short)0, acc[nt], false, false);
    }
  }
}

// ssp(acc) -> bf16 row-major [16 x 64] LDS activation tile
__device__ __forceinline__ void store_act(__bf16* act, const v8f acc[4], int lane) {
  int nl = lane & 15, hi = lane >> 4;
#pragma unroll
  for (int nt = 0; nt < 4; ++nt)
#pragma unroll
    for (int r = 0; r < 8; ++r) {
      int m = r + 8 * hi;
      act[m * EMB + nt * 16 + nl] = (__bf16)sspf(acc[nt][r]);
    }
}

// ---------- kernel 1: hx = mlp(x, hW1, hb1, hW2, hb2) ----------
__global__ __launch_bounds__(128) void hx_kernel(
    const float* __restrict__ x, const float* __restrict__ hW1, const float* __restrict__ hb1,
    const float* __restrict__ hW2, const float* __restrict__ hb2, float* __restrict__ hx) {
  __shared__ v16bf sW1f[256], sW2f[256];
  __shared__ float sb1[EMB], sb2[EMB];
  __shared__ __attribute__((aligned(32))) __bf16 sA[4][16 * EMB];
  __shared__ __attribute__((aligned(32))) __bf16 sH[4][16 * EMB];
  int tid = threadIdx.x, wave = tid >> 5, lane = tid & 31;
  load_w_swz(sW1f, hW1, EMB, tid, 128);
  load_w_swz(sW2f, hW2, EMB, tid, 128);
  load_f32(sb1, hb1, EMB, tid, 128);
  load_f32(sb2, hb2, EMB, tid, 128);
  long row0 = (long)blockIdx.x * 64 + wave * 16;
  stage_bf16(sA[wave], x + row0 * EMB, 16 * EMB, lane);
  __syncthreads();
  v8f acc[4];
  gemm16xK(sA[wave], EMB, sW1f, sb1, EMB, lane, acc);
  store_act(sH[wave], acc, lane);
  __syncthreads();
  gemm16xK(sH[wave], EMB, sW2f, sb2, EMB, lane, acc);
  int nl = lane & 15, hi = lane >> 4;
#pragma unroll
  for (int nt = 0; nt < 4; ++nt)
#pragma unroll
    for (int r = 0; r < 8; ++r)
      hx[(row0 + r + 8 * hi) * EMB + nt * 16 + nl] = acc[nt][r];
}

// ---------- kernel 2: nuclear messages + z_nuc ----------
__global__ __launch_bounds__(128) void nuc_kernel(
    const float* __restrict__ edges_nuc, const float* __restrict__ Y,
    const float* __restrict__ wW1, const float* __restrict__ wb1,
    const float* __restrict__ wW2, const float* __restrict__ wb2,
    float* __restrict__ msg_nuc, float* __restrict__ z_nuc) {
  __shared__ v16bf sW1f[128], sW2f[256];
  __shared__ float sb1[EMB], sb2[EMB];
  __shared__ float sY[NNUC * EMB];
  __shared__ __attribute__((aligned(32))) __bf16 sE[4][NNUC * DIST];
  __shared__ __attribute__((aligned(32))) __bf16 sH[4][NNUC * EMB];
  int tid = threadIdx.x, wave = tid >> 5, lane = tid & 31;
  int b = blockIdx.x >> 4;
  int i = (blockIdx.x & 15) * 4 + wave;
  if (wave < 2)  // Y tile is f32: use async DMA to LDS (512 floats per wave)
    async_copy_f32(sY + wave * 512, Y + (long)b * NNUC * EMB + wave * 512, 512, lane);
  load_w_swz(sW1f, wW1 + 2 * DIST * EMB, DIST, tid, 128);  // l = 2 ("n")
  load_w_swz(sW2f, wW2 + 2 * EMB * EMB, EMB, tid, 128);
  load_f32(sb1, wb1 + 2 * EMB, EMB, tid, 128);
  load_f32(sb2, wb2 + 2 * EMB, EMB, tid, 128);
  stage_bf16(sE[wave], edges_nuc + ((long)(b * NELEC + i) * NNUC) * DIST,
             NNUC * DIST, lane);
  async_wait0();   // sY visible before the block-wide barrier
  __syncthreads();
  v8f acc[4];
  gemm16xK(sE[wave], DIST, sW1f, sb1, DIST, lane, acc);
  store_act(sH[wave], acc, lane);
  __syncthreads();
  gemm16xK(sH[wave], EMB, sW2f, sb2, EMB, lane, acc);
  int nl = lane & 15, hi = lane >> 4;
#pragma unroll
  for (int nt = 0; nt < 4; ++nt) {
    float colsum = 0.0f;
#pragma unroll
    for (int r = 0; r < 8; ++r) {
      int a = r + 8 * hi;  // nucleus index
      float v = acc[nt][r] * sY[a * EMB + nt * 16 + nl];
      msg_nuc[((long)(b * NELEC + i) * NNUC + a) * EMB + nt * 16 + nl] = v;
      colsum += v;
    }
    colsum += __shfl_xor(colsum, 16, 32);
    if (hi == 0) z_nuc[(long)(b * NELEC + i) * EMB + nt * 16 + nl] = colsum;
  }
}

// ---------- kernel 3: electron messages + z_same/z_anti ----------
__global__ __launch_bounds__(128) void elec_kernel(
    const float* __restrict__ edges, const float* __restrict__ wW1,
    const float* __restrict__ wb1, const float* __restrict__ wW2,
    const float* __restrict__ wb2, const float* __restrict__ hx,
    float* __restrict__ msg_el, float* __restrict__ z_same, float* __restrict__ z_anti) {
  __shared__ v16bf sW1f[2][128], sW2f[2][256];
  __shared__ float sb1[2][EMB], sb2[2][EMB];
  __shared__ __attribute__((aligned(32))) __bf16 sE[4][16 * DIST];
  __shared__ __attribute__((aligned(32))) __bf16 sH[4][16 * EMB];
  __shared__ float sHx[4][16 * EMB];
  __shared__ float sz[4][EMB];  // per-wave column partials
  int tid = threadIdx.x, wave = tid >> 5, lane = tid & 31;
  int b = blockIdx.x >> 6, i = blockIdx.x & 63;
  int j0 = wave * 16;
  // hx tile is f32 and wave-private: async DMA straight to LDS, drained late.
  async_copy_f32(sHx[wave], hx + (long)(b * NELEC + j0) * EMB, 16 * EMB, lane);
  load_w_swz(&sW1f[0][0], wW1, DIST, tid, 128);                  // l=0 (same)
  load_w_swz(&sW1f[1][0], wW1 + DIST * EMB, DIST, tid, 128);     // l=1 (anti)
  load_w_swz(&sW2f[0][0], wW2, EMB, tid, 128);
  load_w_swz(&sW2f[1][0], wW2 + EMB * EMB, EMB, tid, 128);
  load_f32(&sb1[0][0], wb1, 2 * EMB, tid, 128);
  load_f32(&sb2[0][0], wb2, 2 * EMB, tid, 128);
  stage_bf16(sE[wave], edges + ((long)blockIdx.x * NELEC + j0) * DIST, 16 * DIST, lane);
  __syncthreads();
  int l = ((i < NUP) == (j0 < NUP)) ? 0 : 1;  // spin-homogeneous 16-row tile
  v8f acc[4];
  gemm16xK(sE[wave], DIST, &sW1f[l][0], &sb1[l][0], DIST, lane, acc);
  store_act(sH[wave], acc, lane);
  __syncthreads();
  gemm16xK(sH[wave], EMB, &sW2f[l][0], &sb2[l][0], EMB, lane, acc);
  async_wait0();  // own wave's sHx tile ready (overlapped with both GEMMs)
  int nl = lane & 15, hi = lane >> 4;
#pragma unroll
  for (int nt = 0; nt < 4; ++nt) {
    float colsum = 0.0f;
#pragma unroll
    for (int r = 0; r < 8; ++r) {
      int m = r + 8 * hi;
      int j = j0 + m;
      float v = acc[nt][r] * sHx[wave][m * EMB + nt * 16 + nl];
      if (j != i) {
        colsum += v;
        int jj = j - ((j > i) ? 1 : 0);  // compacted pair index (uu|ud / du|dd order)
        msg_el[((long)blockIdx.x * (NELEC - 1) + jj) * EMB + nt * 16 + nl] = v;
      }
    }
    colsum += __shfl_xor(colsum, 16, 32);
    if (hi == 0) sz[wave][nt * 16 + nl] = colsum;
  }
  __syncthreads();
  int sameBase = (i < NUP) ? 0 : 2;  // waves {0,1} = up-j tiles, {2,3} = down-j tiles
  int antiBase = 2 - sameBase;
  if (tid < EMB) {
    z_same[(long)blockIdx.x * EMB + tid] = sz[sameBase][tid] + sz[sameBase + 1][tid];
  } else {
    int t = tid - EMB;
    z_anti[(long)blockIdx.x * EMB + t] = sz[antiBase][t] + sz[antiBase + 1][t];
  }
}

// ---------- kernel 4: update = g0(z_same) + g1(z_anti) + g2(z_nuc) ----------
__global__ __launch_bounds__(128) void update_kernel(
    const float* __restrict__ z_same, const float* __restrict__ z_anti,
    const float* __restrict__ z_nuc, const float* __restrict__ gW1,
    const float* __restrict__ gb1, const float* __restrict__ gW2,
    const float* __restrict__ gb2, float* __restrict__ out) {
  __shared__ v16bf sW1f[3][256], sW2f[3][256];
  __shared__ float sb1[3][EMB], sb2[3][EMB];
  __shared__ __attribute__((aligned(32))) __bf16 sA[4][16 * EMB];
  __shared__ __attribute__((aligned(32))) __bf16 sH[4][16 * EMB];
  int tid = threadIdx.x, wave = tid >> 5, lane = tid & 31;
#pragma unroll
  for (int l = 0; l < 3; ++l) {
    load_w_swz(&sW1f[l][0], gW1 + l * EMB * EMB, EMB, tid, 128);
    load_w_swz(&sW2f[l][0], gW2 + l * EMB * EMB, EMB, tid, 128);
  }
  load_f32(&sb1[0][0], gb1, 3 * EMB, tid, 128);
  load_f32(&sb2[0][0], gb2, 3 * EMB, tid, 128);
  long row0 = (long)blockIdx.x * 64 + wave * 16;
  const float* zs[3] = {z_same + row0 * EMB, z_anti + row0 * EMB, z_nuc + row0 * EMB};
  float oacc[4][8];
#pragma unroll
  for (int nt = 0; nt < 4; ++nt)
#pragma unroll
    for (int r = 0; r < 8; ++r) oacc[nt][r] = 0.0f;
  for (int l = 0; l < 3; ++l) {
    __syncthreads();  // protect sA/sH reuse across iterations + weight load on l==0
    stage_bf16(sA[wave], zs[l], 16 * EMB, lane);
    __syncthreads();
    v8f acc[4];
    gemm16xK(sA[wave], EMB, &sW1f[l][0], &sb1[l][0], EMB, lane, acc);
    store_act(sH[wave], acc, lane);
    __syncthreads();
    gemm16xK(sH[wave], EMB, &sW2f[l][0], &sb2[l][0], EMB, lane, acc);
#pragma unroll
    for (int nt = 0; nt < 4; ++nt)
#pragma unroll
      for (int r = 0; r < 8; ++r) oacc[nt][r] += acc[nt][r];
  }
  int nl = lane & 15, hi = lane >> 4;
#pragma unroll
  for (int nt = 0; nt < 4; ++nt)
#pragma unroll
    for (int r = 0; r < 8; ++r)
      out[(row0 + r + 8 * hi) * EMB + nt * 16 + nl] = oacc[nt][r];
}

// ---------- launch ----------
extern "C" void kernel_launch(void* const* d_in, const int* in_sizes, int n_in,
                              void* d_out, int out_size, void* d_ws, size_t ws_size,
                              hipStream_t stream) {
  (void)in_sizes; (void)n_in; (void)out_size; (void)ws_size;
  const float* x   = (const float*)d_in[0];
  const float* Y   = (const float*)d_in[1];
  const float* ee  = (const float*)d_in[2];
  const float* en  = (const float*)d_in[3];
  const float* wW1 = (const float*)d_in[4];
  const float* wb1 = (const float*)d_in[5];
  const float* wW2 = (const float*)d_in[6];
  const float* wb2 = (const float*)d_in[7];
  const float* hW1 = (const float*)d_in[8];
  const float* hb1 = (const float*)d_in[9];
  const float* hW2 = (const float*)d_in[10];
  const float* hb2 = (const float*)d_in[11];
  const float* gW1 = (const float*)d_in[12];
  const float* gb1 = (const float*)d_in[13];
  const float* gW2 = (const float*)d_in[14];
  const float* gb2 = (const float*)d_in[15];

  const long NROWS = (long)B_ * NELEC;  // 16384
  float* ws     = (float*)d_ws;
  float* hx     = ws;                    // [16384, 64]
  float* z_same = ws + NROWS * EMB;      // [16384, 64]
  float* z_anti = ws + 2 * NROWS * EMB;
  float* z_nuc  = ws + 3 * NROWS * EMB;

  float* out_update = (float*)d_out;                           // [256,64,64]
  float* out_msg_el = out_update + NROWS * EMB;                // [256,64,63,64]
  float* out_msg_nu = out_msg_el + NROWS * (NELEC - 1) * EMB;  // [256,64,16,64]

  hx_kernel<<<B_ * NELEC / 64, 128, 0, stream>>>(x, hW1, hb1, hW2, hb2, hx);
  nuc_kernel<<<B_ * (NELEC / 4), 128, 0, stream>>>(en, Y, wW1, wb1, wW2, wb2,
                                                   out_msg_nu, z_nuc);
  elec_kernel<<<B_ * NELEC, 128, 0, stream>>>(ee, wW1, wb1, wW2, wb2, hx,
                                              out_msg_el, z_same, z_anti);
  update_kernel<<<B_ * NELEC / 64, 128, 0, stream>>>(z_same, z_anti, z_nuc,
                                                     gW1, gb1, gW2, gb2, out_update);
}
